// PointInterpDecoder_39213051412833
// MI455X (gfx1250) — matrix operand order repaired and checked by reading, not compiled, not run on hardware
//
#include <hip/hip_runtime.h>
#include <hip/hip_bf16.h>

// ---------------------------------------------------------------------------
// PointInterpDecoder fused CDNA5 kernel (round 3).
// Stage 1: Gaussian-softmax interpolation as f16 WMMA GEMM, 32-query M-tile,
//          software-pipelined (double-buffered weights/anchors, 1 barrier/chunk),
//          raw v_sqrt_f32/v_exp_f32 for the weight math,
//          global_prefetch_b8 on next feats chunk.
// Stage 2: full MLP (fc0 -> fc1 -> 5 resnet blocks -> out) via f16 WMMA,
//          f32 accumulation, activations staged in LDS.
// ---------------------------------------------------------------------------

typedef _Float16 v16h __attribute__((ext_vector_type(16)));
typedef _Float16 v8h  __attribute__((ext_vector_type(8)));
typedef _Float16 v4h  __attribute__((ext_vector_type(4)));
typedef float    v8f  __attribute__((ext_vector_type(8)));

#define B_    4
#define NQ_   8192
#define NP_   2048
#define DIN_  256     // dim_inp
#define DIM_  200     // dim (valid)
#define DIMP_ 224     // dim padded to multiple of 32 for WMMA K
#define HID_  128
#define OUT_  3
#define NBLK_ 5
#define MT_   32      // queries per workgroup (two 16-row WMMA M tiles)

// ---------------------------------------------------------------------------
// Fragment helpers (wave32, V_WMMA_F32_16X16X32_F16 layouts per CDNA5 ISA 7.12.2)
// A (16x32 f16): lane l holds row m=l&15; VGPR0-3 = K [kb,kb+8), VGPR4-7 =
// K [kb+16,kb+24) with kb = (l>=16)?8:0.  Row-major LDS rows -> 2x b128 loads.
// B (32x16 f16): lane l holds col n=l&15; same K order as A (kb, kb+16).
// [N][K] weight layout -> 2x b128 loads.
// C/D (16x16 f32): lane l holds col n=l&15, VGPR r = row (l>=16?8:0)+r.
// ---------------------------------------------------------------------------

__device__ __forceinline__ v16h load_a_frag(const _Float16* A, int rowStride,
                                            int k0, int lane, bool doRelu) {
    int m  = lane & 15;
    int kb = (lane >> 4) * 8;
    const _Float16* p = A + m * rowStride + k0 + kb;
    v8h lo = *(const v8h*)(p);
    v8h hi = *(const v8h*)(p + 16);
    v16h a = __builtin_shufflevector(lo, hi, 0,1,2,3,4,5,6,7,8,9,10,11,12,13,14,15);
    if (doRelu) {
#pragma unroll
        for (int i = 0; i < 16; ++i)
            a[i] = a[i] > (_Float16)0.f ? a[i] : (_Float16)0.f;
    }
    return a;
}

__device__ __forceinline__ v16h load_b_frag(const _Float16* Wt, int Kpad,
                                            int k0, int lane) {
    const _Float16* p = Wt + (size_t)(lane & 15) * Kpad + k0 + (lane >> 4) * 8;
    v8h lo = *(const v8h*)(p);
    v8h hi = *(const v8h*)(p + 16);
    return __builtin_shufflevector(lo, hi, 0,1,2,3,4,5,6,7,8,9,10,11,12,13,14,15);
}

__device__ __forceinline__ v8f wmma_f16(v16h a, v16h b, v8f c) {
    return __builtin_amdgcn_wmma_f32_16x16x32_f16(false, a, false, b,
                                                  (short)0, c, false, false);
}

__device__ __forceinline__ v8f gemm16(const _Float16* ldsA, int aStride, bool reluA,
                                      const _Float16* gW, int Kpad, int nChunks,
                                      int lane, v8f acc) {
    for (int kc = 0; kc < nChunks; ++kc) {
        v16h a = load_a_frag(ldsA, aStride, kc * 32, lane, reluA);
        v16h b = load_b_frag(gW, Kpad, kc * 32, lane);
        acc = wmma_f16(a, b, acc);
    }
    return acc;
}

__device__ __forceinline__ v8f add_bias(v8f acc, const float* bias, int nbase,
                                        int nvalid, int lane) {
    int n = nbase + (lane & 15);
    float bv = (n < nvalid) ? bias[n] : 0.f;
#pragma unroll
    for (int r = 0; r < 8; ++r) acc[r] += bv;
    return acc;
}

__device__ __forceinline__ void store_c_lds(_Float16* buf, int rowStride, int ntile,
                                            v8f acc, bool doRelu, int lane) {
    int n  = ntile * 16 + (lane & 15);
    int mb = (lane >> 4) * 8;
#pragma unroll
    for (int r = 0; r < 8; ++r) {
        float v = acc[r];
        if (doRelu) v = v > 0.f ? v : 0.f;
        buf[(mb + r) * rowStride + n] = (_Float16)v;
    }
}

// exp-weight tile: thread owns (query q, 4 anchors p4*4..p4*4+3) of one chunk.
// logits <= 0 so exp never overflows; max-subtraction pass is unnecessary.
// Raw v_sqrt_f32 / v_exp_f32: result is squared immediately (sqrt) and fed to
// f16 (exp), so the precise-sqrt fixup sequence is pure overhead.
#define NEG25_LOG2E (-25.0f * 1.44269504088896340736f)   // -log2(e)/VAR

__device__ __forceinline__ float calc_weights(const float* ax, _Float16* swn,
                                              float qx, float qy, float qz,
                                              int q, int p4) {
    float s = 0.f;
    _Float16 h[4];
#pragma unroll
    for (int j = 0; j < 4; ++j) {
        int a = p4 * 4 + j;
        float dx = ax[a * 3 + 0] - qx;
        float dy = ax[a * 3 + 1] - qy;
        float dz = ax[a * 3 + 2] - qz;
        float d2 = dx * dx + dy * dy + dz * dz;
        float d  = __builtin_amdgcn_sqrtf(d2) + 1e-5f;   // v_sqrt_f32
        float w  = __builtin_amdgcn_exp2f(d * d * NEG25_LOG2E); // v_exp_f32
        s += w;
        h[j] = (_Float16)w;
    }
    *(v4h*)(&swn[q * 32 + p4 * 4]) = (v4h){h[0], h[1], h[2], h[3]};  // ds_store_b64
    return s;
}

// ---------------------------------------------------------------------------
// Prep kernels: f32 -> f16, transpose to fragment-friendly [N][K] (padded).
// ---------------------------------------------------------------------------
__global__ void k_featsT(const float* __restrict__ in, _Float16* __restrict__ out) {
    size_t idx = (size_t)blockIdx.x * 256 + threadIdx.x;   // [B][DIN][NP]
    if (idx >= (size_t)B_ * DIN_ * NP_) return;
    int a = (int)(idx % NP_);
    int r = (int)(idx / NP_);
    int n = r % DIN_;
    int b = r / DIN_;
    out[idx] = (_Float16)in[((size_t)b * NP_ + a) * DIN_ + n];
}

__global__ void k_wT(const float* __restrict__ in, _Float16* __restrict__ out,
                     int K, int N, int Kpad, int Npad) {
    int idx = blockIdx.x * 256 + threadIdx.x;              // [Npad][Kpad]
    if (idx >= Npad * Kpad) return;
    int k = idx % Kpad, n = idx / Kpad;
    out[idx] = (k < K && n < N) ? (_Float16)in[(size_t)k * N + n] : (_Float16)0.f;
}

// ---------------------------------------------------------------------------
// Fused kernel: one 32-query tile per workgroup, 8 wave32s.
// ---------------------------------------------------------------------------
__global__ __launch_bounds__(256) void fused_decoder(
    const float* __restrict__ xyz_q,       // [B][NQ][3]
    const float* __restrict__ anchors,     // [B][NP][3]
    const _Float16* __restrict__ featsT,   // [B][DIN][NP] f16
    const _Float16* __restrict__ fc0T,     // [DIMP][DIN]
    const float* __restrict__ fc0_b,       // [DIM]
    const _Float16* __restrict__ fc1T,     // [HID][DIMP]
    const float* __restrict__ fc1_b,       // [HID]
    const _Float16* __restrict__ w0T,      // [NBLK][HID][HID]
    const float* __restrict__ blk_b0,      // [NBLK][HID]
    const _Float16* __restrict__ w1T,      // [NBLK][HID][HID]
    const float* __restrict__ blk_b1,      // [NBLK][HID]
    const _Float16* __restrict__ fccT,     // [NBLK][HID][DIMP]
    const float* __restrict__ fcc_b,       // [NBLK][HID]
    const _Float16* __restrict__ outT,     // [16][HID]
    const float* __restrict__ out_b,       // [OUT]
    float* __restrict__ out)               // [B][NQ][3]
{
    __shared__ float    sQ[MT_ * 3];          // 32 query xyz
    __shared__ float    sAx[2][96];           // double-buffered anchor xyz
    __shared__ _Float16 sW[2][MT_ * 32];      // double-buffered weight tiles
    __shared__ float    sWpart[256];          // per-thread weight-sum partials
    __shared__ float    sRecip[MT_];          // 1/sum per query
    __shared__ _Float16 sC[MT_ * DIN_];       // interpolated features (f16)
    __shared__ _Float16 sLat[MT_ * DIMP_];    // lat (pre-relu, padded)
    __shared__ _Float16 sNet[MT_ * HID_];     // relu(net)
    __shared__ _Float16 sT[MT_ * HID_];       // relu(block hidden)

    const int tile = blockIdx.x;
    const int b    = tile / (NQ_ / MT_);
    const int q0   = (tile % (NQ_ / MT_)) * MT_;
    const int t    = threadIdx.x;
    const int lane = t & 31;
    const int wave = t >> 5;
    const int q    = t & 31;   // query slot for weight calc (0..31)
    const int p4   = t >> 5;   // anchor-quad slot (0..7)

    if (t < MT_ * 3) sQ[t] = xyz_q[((size_t)b * NQ_ + q0) * 3 + t];
    if (t < 96) sAx[0][t] = anchors[((size_t)b * NP_) * 3 + t];
    __syncthreads();

    const float qx = sQ[q * 3 + 0];
    const float qy = sQ[q * 3 + 1];
    const float qz = sQ[q * 3 + 2];

    // ---------------- Stage 1: softmax-weighted interpolation -------------
    // Pipeline prologue: weights for chunk 0, anchors for chunk 1.
    float wsum = calc_weights(sAx[0], sW[0], qx, qy, qz, q, p4);
    if (t < 96) sAx[1][t] = anchors[((size_t)b * NP_ + 32) * 3 + t];
    __syncthreads();

    v8f cacc[2][2] = {};   // [m-half][feature tile]
    const int NCHUNK = NP_ / 32;

    for (int kc = 0; kc < NCHUNK; ++kc) {
        const int cur = kc & 1;
        const _Float16* swc = sW[cur];
        const _Float16* fb  = featsT + (size_t)(b * DIN_ + wave * 32) * NP_
                                     + (size_t)kc * 32;

        // 2 A frags (query halves) x 2 B frags (feature tiles) -> 4 WMMAs
        v16h a0 = load_a_frag(swc,            32, 0, lane, false);
        v16h a1 = load_a_frag(swc + 16 * 32,  32, 0, lane, false);
        v16h b0 = load_b_frag(fb,                    NP_, 0, lane);
        v16h b1 = load_b_frag(fb + (size_t)16 * NP_, NP_, 0, lane);
        cacc[0][0] = wmma_f16(a0, b0, cacc[0][0]);
        cacc[1][0] = wmma_f16(a1, b0, cacc[1][0]);
        cacc[0][1] = wmma_f16(a0, b1, cacc[0][1]);
        cacc[1][1] = wmma_f16(a1, b1, cacc[1][1]);

        if (kc + 1 < NCHUNK) {
            // warm L2/WGP for next chunk's B fragments (global_prefetch_b8)
            __builtin_prefetch(fb + 32, 0, 1);
            __builtin_prefetch(fb + (size_t)16 * NP_ + 32, 0, 1);
            // exp-weights for chunk kc+1 (co-executes with WMMA pipe)
            wsum += calc_weights(sAx[(kc + 1) & 1], sW[(kc + 1) & 1],
                                 qx, qy, qz, q, p4);
        }
        if (kc + 2 < NCHUNK && t < 96)
            sAx[kc & 1][t] = anchors[((size_t)b * NP_ + (kc + 2) * 32) * 3 + t];
        __syncthreads();   // publish sW[next] / sAx[next]; sW[cur] reads done
    }

    // reduce weight sums (8 partials per query) and normalize
    sWpart[p4 * 32 + q] = wsum;
    __syncthreads();
    if (t < MT_) {
        float s = 0.f;
#pragma unroll
        for (int j = 0; j < 8; ++j) s += sWpart[j * 32 + t];
        sRecip[t] = 1.0f / s;
    }
    __syncthreads();
#pragma unroll
    for (int m2 = 0; m2 < 2; ++m2) {
        int mb = m2 * 16 + (lane >> 4) * 8;
        float rc[8];
#pragma unroll
        for (int r = 0; r < 8; ++r) rc[r] = sRecip[mb + r];
#pragma unroll
        for (int nt = 0; nt < 2; ++nt) {
#pragma unroll
            for (int r = 0; r < 8; ++r) cacc[m2][nt][r] *= rc[r];
            store_c_lds(sC + m2 * 16 * DIN_, DIN_, wave * 2 + nt,
                        cacc[m2][nt], false, lane);
        }
    }
    __syncthreads();

    // ---------------- Stage 2: MLP ----------------------------------------
    // lat = c @ fc0 + b0  (N padded to 224; rows 200..223 are exact zeros)
    for (int nt = wave; nt < DIMP_ / 16; nt += 8) {
#pragma unroll
        for (int m2 = 0; m2 < 2; ++m2) {
            v8f acc = {};
            acc = gemm16(sC + m2 * 16 * DIN_, DIN_, false,
                         fc0T + (size_t)nt * 16 * DIN_, DIN_, DIN_ / 32, lane, acc);
            acc = add_bias(acc, fc0_b, nt * 16, DIM_, lane);
            store_c_lds(sLat + m2 * 16 * DIMP_, DIMP_, nt, acc, false, lane);
        }
    }
    __syncthreads();

    // net = relu(lat) @ fc1 + b1; each wave owns one N-tile for both M halves
    v8f net[2];
#pragma unroll
    for (int m2 = 0; m2 < 2; ++m2) {
        v8f acc = {};
        acc = gemm16(sLat + m2 * 16 * DIMP_, DIMP_, true,
                     fc1T + (size_t)wave * 16 * DIMP_, DIMP_, DIMP_ / 32, lane, acc);
        net[m2] = add_bias(acc, fc1_b, wave * 16, HID_, lane);
    }

    for (int i = 0; i < NBLK_; ++i) {
        // net += lat @ fcc[i] + fcc_b[i]  (accumulate into live C fragments)
#pragma unroll
        for (int m2 = 0; m2 < 2; ++m2) {
            net[m2] = gemm16(sLat + m2 * 16 * DIMP_, DIMP_, false,
                             fccT + (size_t)(i * HID_ + wave * 16) * DIMP_,
                             DIMP_, DIMP_ / 32, lane, net[m2]);
            net[m2] = add_bias(net[m2], fcc_b + i * HID_, wave * 16, HID_, lane);
            store_c_lds(sNet + m2 * 16 * HID_, HID_, wave, net[m2], true, lane);
        }
        __syncthreads();

#pragma unroll
        for (int m2 = 0; m2 < 2; ++m2) {
            v8f tacc = {};
            tacc = gemm16(sNet + m2 * 16 * HID_, HID_, false,
                          w0T + (size_t)(i * HID_ + wave * 16) * HID_,
                          HID_, HID_ / 32, lane, tacc);
            tacc = add_bias(tacc, blk_b0 + i * HID_, wave * 16, HID_, lane);
            store_c_lds(sT + m2 * 16 * HID_, HID_, wave, tacc, true, lane);
        }
        __syncthreads();

        // net += t @ w1[i] + b1[i]
#pragma unroll
        for (int m2 = 0; m2 < 2; ++m2) {
            net[m2] = gemm16(sT + m2 * 16 * HID_, HID_, false,
                             w1T + (size_t)(i * HID_ + wave * 16) * HID_,
                             HID_, HID_ / 32, lane, net[m2]);
            net[m2] = add_bias(net[m2], blk_b1 + i * HID_, wave * 16, HID_, lane);
        }
        __syncthreads();   // all reads of sNet/sT done before next rewrite
    }

    // out = relu(net) @ out_w + out_b
#pragma unroll
    for (int m2 = 0; m2 < 2; ++m2)
        store_c_lds(sNet + m2 * 16 * HID_, HID_, wave, net[m2], true, lane);
    __syncthreads();
    if (wave < 2) {
        const int m2 = wave;
        v8f oacc = {};
        oacc = gemm16(sNet + m2 * 16 * HID_, HID_, false, outT, HID_,
                      HID_ / 32, lane, oacc);
        int n  = lane & 15;
        int mb = (lane >> 4) * 8;
        if (n < OUT_) {
            float bv = out_b[n];
#pragma unroll
            for (int r = 0; r < 8; ++r)
                out[((size_t)b * NQ_ + q0 + m2 * 16 + mb + r) * 3 + n] = oacc[r] + bv;
        }
    }
}

// ---------------------------------------------------------------------------
extern "C" void kernel_launch(void* const* d_in, const int* in_sizes, int n_in,
                              void* d_out, int out_size, void* d_ws, size_t ws_size,
                              hipStream_t stream) {
    const float* xyz_q        = (const float*)d_in[0];
    const float* anchors      = (const float*)d_in[1];
    const float* anchor_feats = (const float*)d_in[2];
    const float* fc0_w  = (const float*)d_in[3];
    const float* fc0_b  = (const float*)d_in[4];
    const float* fc1_w  = (const float*)d_in[5];
    const float* fc1_b  = (const float*)d_in[6];
    const float* blk_w0 = (const float*)d_in[7];
    const float* blk_b0 = (const float*)d_in[8];
    const float* blk_w1 = (const float*)d_in[9];
    const float* blk_b1 = (const float*)d_in[10];
    const float* fcc_w  = (const float*)d_in[11];
    const float* fcc_b  = (const float*)d_in[12];
    const float* out_w  = (const float*)d_in[13];
    const float* out_b  = (const float*)d_in[14];
    float* out = (float*)d_out;

    // workspace carve-up (f16 elements)
    _Float16* featsT = (_Float16*)d_ws;                       // B*DIN*NP
    _Float16* fc0T   = featsT + (size_t)B_ * DIN_ * NP_;      // DIMP*DIN
    _Float16* fc1T   = fc0T   + (size_t)DIMP_ * DIN_;         // HID*DIMP
    _Float16* fccT   = fc1T   + (size_t)HID_ * DIMP_;         // NBLK*HID*DIMP
    _Float16* w0T    = fccT   + (size_t)NBLK_ * HID_ * DIMP_; // NBLK*HID*HID
    _Float16* w1T    = w0T    + (size_t)NBLK_ * HID_ * HID_;  // NBLK*HID*HID
    _Float16* outT   = w1T    + (size_t)NBLK_ * HID_ * HID_;  // 16*HID

    // ---- prep: convert/transpose/pad everything to f16 fragment layouts ----
    {
        size_t nft = (size_t)B_ * DIN_ * NP_;
        k_featsT<<<(unsigned)((nft + 255) / 256), 256, 0, stream>>>(anchor_feats, featsT);
    }
    k_wT<<<(DIMP_ * DIN_ + 255) / 256, 256, 0, stream>>>(fc0_w, fc0T, DIN_, DIM_, DIN_, DIMP_);
    k_wT<<<(HID_ * DIMP_ + 255) / 256, 256, 0, stream>>>(fc1_w, fc1T, DIM_, HID_, DIMP_, HID_);
    for (int i = 0; i < NBLK_; ++i) {
        k_wT<<<(HID_ * DIMP_ + 255) / 256, 256, 0, stream>>>(
            fcc_w + (size_t)i * DIM_ * HID_, fccT + (size_t)i * HID_ * DIMP_,
            DIM_, HID_, DIMP_, HID_);
        k_wT<<<(HID_ * HID_ + 255) / 256, 256, 0, stream>>>(
            blk_w0 + (size_t)i * HID_ * HID_, w0T + (size_t)i * HID_ * HID_,
            HID_, HID_, HID_, HID_);
        k_wT<<<(HID_ * HID_ + 255) / 256, 256, 0, stream>>>(
            blk_w1 + (size_t)i * HID_ * HID_, w1T + (size_t)i * HID_ * HID_,
            HID_, HID_, HID_, HID_);
    }
    k_wT<<<(16 * HID_ + 255) / 256, 256, 0, stream>>>(out_w, outT, HID_, OUT_, HID_, 16);

    // ---- fused interpolation + MLP ----
    fused_decoder<<<B_ * (NQ_ / MT_), 256, 0, stream>>>(
        xyz_q, anchors, featsT,
        fc0T, fc0_b, fc1T, fc1_b,
        w0T, blk_b0, w1T, blk_b1,
        fccT, fcc_b, outT, out_b, out);
}
